// Attention_56736517980223
// MI455X (gfx1250) — matrix operand order repaired and checked by reading, not compiled, not run on hardware
//
#include <hip/hip_runtime.h>
#include <hip/hip_bf16.h>
#include <stdint.h>

typedef __bf16 bf16;
typedef __attribute__((ext_vector_type(16))) __bf16 v16bf;
typedef __attribute__((ext_vector_type(8)))  __bf16 v8bf;
typedef __attribute__((ext_vector_type(8)))  float   v8f;

#define BATCH 2
#define SEQ   2048
#define DIM   2048
#define NH    32
#define NKV   8
#define HD    64
#define MTOT  (BATCH*SEQ)   // 4096
#define KVD   (NKV*HD)      // 512

#define GM_ROPE 1
#define GM_VT   2
#define GM_F32  3

__device__ __forceinline__ v8f wmma_bf16(v16bf a, v16bf b, v8f c) {
  return __builtin_amdgcn_wmma_f32_16x16x32_bf16(false, a, false, b, (short)0, c,
                                                 false, false);
}
__device__ __forceinline__ v8f vzero8() {
  v8f z;
#pragma unroll
  for (int i = 0; i < 8; ++i) z[i] = 0.0f;
  return z;
}

union AFrag { v16bf v; v8bf h2[2]; uint32_t u[8]; };

// ---------------- precision conversion kernels ----------------
__global__ void cvt_f32_bf16(const float* __restrict__ src, bf16* __restrict__ dst, int n) {
  int i = blockIdx.x * blockDim.x + threadIdx.x;
  if (i < n) dst[i] = (bf16)src[i];
}
// src [K][N] f32  ->  dst [N][K] bf16  (weights transposed for contiguous B-fragments)
__global__ void cvtT_f32_bf16(const float* __restrict__ src, bf16* __restrict__ dst,
                              int K, int N) {
  int i = blockIdx.x * blockDim.x + threadIdx.x;
  if (i < K * N) {
    int k = i / N, n = i - k * N;
    dst[(size_t)n * K + k] = (bf16)src[i];
  }
}

// ---------------- bf16 WMMA GEMM: C[M][N] = A[M][Kd] * WT[N][Kd]^T ----------------
// block = 256 threads = 8 waves; block tile: 32 rows x 512 cols; wave tile 32 x 64.
// Per K-step(32): 8 WMMAs from 4 A-half loads + 8 B b128 loads (1.5 loads/WMMA).
__global__ __launch_bounds__(256)
void gemm_bf16_kernel(const bf16* __restrict__ A, const bf16* __restrict__ WT,
                      void* __restrict__ outp, int Kd, int N, int mode,
                      const float* __restrict__ cosp, const float* __restrict__ sinp)
{
  const int lane = threadIdx.x & 31;
  const int wave = threadIdx.x >> 5;
  const int hf   = lane >> 4;     // wave32 half
  const int l16  = lane & 15;
  const int row0 = blockIdx.x << 5;                 // 32-row block/wave M tile
  const int nbase = (blockIdx.y << 9) + (wave << 6);

  const bf16* aptr0 = A + (size_t)(row0 + l16) * Kd;
  const bf16* aptr1 = A + (size_t)(row0 + 16 + l16) * Kd;
  const bf16* wptr[4];
#pragma unroll
  for (int nt = 0; nt < 4; ++nt)
    wptr[nt] = WT + (size_t)(nbase + nt * 16 + l16) * Kd + hf * 16;

  v8f acc[2][4];
#pragma unroll
  for (int mt = 0; mt < 2; ++mt)
#pragma unroll
    for (int nt = 0; nt < 4; ++nt) acc[mt][nt] = vzero8();

  for (int k = 0; k < Kd; k += 32) {
    AFrag af0, af1; // A layout: elems 0-15 -> K = hf*16+e
    af0.h2[0] = *reinterpret_cast<const v8bf*>(aptr0 + k + hf * 16);
    af0.h2[1] = *reinterpret_cast<const v8bf*>(aptr0 + k + hf * 16 + 8);
    af1.h2[0] = *reinterpret_cast<const v8bf*>(aptr1 + k + hf * 16);
    af1.h2[1] = *reinterpret_cast<const v8bf*>(aptr1 + k + hf * 16 + 8);
#pragma unroll
    for (int nt = 0; nt < 4; ++nt) {
      v16bf bw = *reinterpret_cast<const v16bf*>(wptr[nt] + k); // lane=N, elems=K contig
      acc[0][nt] = wmma_bf16(af0.v, bw, acc[0][nt]);
      acc[1][nt] = wmma_bf16(af1.v, bw, acc[1][nt]);
    }
  }

  if (mode == GM_F32) {
    float* o = (float*)outp;
#pragma unroll
    for (int mt = 0; mt < 2; ++mt)
#pragma unroll
      for (int nt = 0; nt < 4; ++nt) {
        int n = nbase + nt * 16 + l16;
#pragma unroll
        for (int r = 0; r < 8; ++r)
          o[(size_t)(row0 + mt * 16 + r + 8 * hf) * N + n] = acc[mt][nt][r];
      }
  } else if (mode == GM_ROPE) { // fused rotary; lane parity == column parity
    bf16* o = (bf16*)outp;
#pragma unroll
    for (int mt = 0; mt < 2; ++mt)
#pragma unroll
      for (int nt = 0; nt < 4; ++nt) {
        int n = nbase + nt * 16 + l16;
        int pidx = (n & 63) >> 1;
#pragma unroll
        for (int r = 0; r < 8; ++r) {
          int m = row0 + mt * 16 + r + 8 * hf;
          int s = m & (SEQ - 1);
          float v = acc[mt][nt][r];
          float p = __shfl_xor(v, 1, 32);
          float c  = cosp[s * 32 + pidx];
          float sn = sinp[s * 32 + pidx];
          float res = (n & 1) ? (v * c + p * sn) : (v * c - p * sn);
          o[(size_t)m * N + n] = (bf16)res;
        }
      }
  } else { // GM_VT : out[n][m]  (V transposed for contiguous PV B-fragments)
    bf16* o = (bf16*)outp;
#pragma unroll
    for (int mt = 0; mt < 2; ++mt)
#pragma unroll
      for (int nt = 0; nt < 4; ++nt) {
        int n = nbase + nt * 16 + l16;
        v8bf pk;
#pragma unroll
        for (int r = 0; r < 8; ++r) pk[r] = (bf16)acc[mt][nt][r];
        *reinterpret_cast<v8bf*>(o + (size_t)n * MTOT + row0 + mt * 16 + 8 * hf) = pk;
      }
  }
}

// ---------------- flash attention: one wave = 16 q-rows of one (b,h) ----------------
__global__ __launch_bounds__(256)
void attn_fwd_kernel(const bf16* __restrict__ Q, const bf16* __restrict__ K,
                     const bf16* __restrict__ VT, bf16* __restrict__ O)
{
  __shared__ bf16 pbuf[8][16 * 34];   // per-wave P-tile staging (C-layout -> A-layout)
  const int lane = threadIdx.x & 31;
  const int wave = threadIdx.x >> 5;
  const int hf   = lane >> 4;
  const int l16  = lane & 15;

  const int gw    = blockIdx.x * 8 + wave;
  const int qtile = gw & (SEQ / 16 - 1);
  const int bh    = gw >> 7;
  const int h     = bh & (NH - 1);
  const int b     = bh >> 5;
  const int kvh   = h >> 2;           // GQA: 4 q-heads per kv-head
  const int qbase = qtile << 4;

  // Q A-fragments for head_dim 64 (two K=32 steps)
  const bf16* qrow = Q + (size_t)(b * SEQ + qbase + l16) * DIM + h * HD;
  AFrag qf[2];
#pragma unroll
  for (int ks = 0; ks < 2; ++ks) {
    qf[ks].h2[0] = *reinterpret_cast<const v8bf*>(qrow + ks * 32 + hf * 16);
    qf[ks].h2[1] = *reinterpret_cast<const v8bf*>(qrow + ks * 32 + hf * 16 + 8);
  }

  v8f acc[4];
#pragma unroll
  for (int nt = 0; nt < 4; ++nt) acc[nt] = vzero8();
  float mrow[8], lrow[8];
#pragma unroll
  for (int r = 0; r < 8; ++r) { mrow[r] = -3.0e38f; lrow[r] = 0.0f; }

  bf16* pb = &pbuf[wave][0];
  const float scale = 0.125f;         // 1/sqrt(64)

  const int kvend = qbase + 16;       // causal bound (exclusive)
  for (int kv = 0; kv < kvend; kv += 32) {
    // ---- scores: two 16x16 tiles over 32 kv positions ----
    v8f st[2];
#pragma unroll
    for (int nt = 0; nt < 2; ++nt) {
      const bf16* kp = K + (size_t)(b * SEQ + kv + nt * 16 + l16) * KVD
                         + kvh * HD + hf * 16;
      v16bf bk0 = *reinterpret_cast<const v16bf*>(kp);       // hd 0..31 slice
      v16bf bk1 = *reinterpret_cast<const v16bf*>(kp + 32);  // hd 32..63 slice
      v8f c = vzero8();
      c = wmma_bf16(qf[0].v, bk0, c);
      c = wmma_bf16(qf[1].v, bk1, c);
      st[nt] = c;
    }

    // ---- online softmax (per-row reductions across the 16-lane half) ----
    float pv0[8], pv1[8];
#pragma unroll
    for (int r = 0; r < 8; ++r) {
      int qpos = qbase + r + 8 * hf;
      float s0 = st[0][r] * scale;
      float s1 = st[1][r] * scale;
      if (kv + l16 > qpos)      s0 = -1.0e30f;
      if (kv + 16 + l16 > qpos) s1 = -1.0e30f;
      float t = fmaxf(s0, s1);
      t = fmaxf(t, __shfl_xor(t, 1, 32));
      t = fmaxf(t, __shfl_xor(t, 2, 32));
      t = fmaxf(t, __shfl_xor(t, 4, 32));
      t = fmaxf(t, __shfl_xor(t, 8, 32));
      float nm = fmaxf(mrow[r], t);
      float p0 = __expf(s0 - nm);
      float p1 = __expf(s1 - nm);
      float rs = p0 + p1;
      rs += __shfl_xor(rs, 1, 32);
      rs += __shfl_xor(rs, 2, 32);
      rs += __shfl_xor(rs, 4, 32);
      rs += __shfl_xor(rs, 8, 32);
      float corr = __expf(mrow[r] - nm);
      lrow[r] = lrow[r] * corr + rs;
      mrow[r] = nm;
#pragma unroll
      for (int nt = 0; nt < 4; ++nt) acc[nt][r] *= corr;
      pv0[r] = p0; pv1[r] = p1;
    }

    // ---- P tile (16x32) C-layout -> LDS -> A-fragment layout ----
#pragma unroll
    for (int r = 0; r < 8; ++r) {
      pb[(r + 8 * hf) * 34 + l16]      = (bf16)pv0[r];
      pb[(r + 8 * hf) * 34 + 16 + l16] = (bf16)pv1[r];
    }
    __builtin_amdgcn_wave_barrier();  // wave-internal; DS ops are in-order per wave
    AFrag pf;
    const bf16* prow = pb + l16 * 34; // m = l16
#pragma unroll
    for (int j = 0; j < 4; ++j) {
      pf.u[j]     = *reinterpret_cast<const uint32_t*>(prow + hf * 16 + 2 * j);
      pf.u[4 + j] = *reinterpret_cast<const uint32_t*>(prow + hf * 16 + 8 + 2 * j);
    }
    __builtin_amdgcn_wave_barrier();

    // ---- PV: V^T layout gives contiguous B-fragments (lane=hd col, elems=kv) ----
#pragma unroll
    for (int nt = 0; nt < 4; ++nt) {
      const bf16* vp = VT + (size_t)(kvh * HD + nt * 16 + l16) * MTOT
                          + b * SEQ + kv + hf * 16;
      v16bf bv = *reinterpret_cast<const v16bf*>(vp);
      acc[nt] = wmma_bf16(pf.v, bv, acc[nt]);
    }
  }

  // ---- normalize + store bf16 attention output [b][tok][h*64+hd] ----
#pragma unroll
  for (int nt = 0; nt < 4; ++nt) {
#pragma unroll
    for (int r = 0; r < 8; ++r) {
      int m = qbase + r + 8 * hf;
      int n = h * HD + nt * 16 + l16;
      O[(size_t)(b * SEQ + m) * DIM + n] = (bf16)(acc[nt][r] / lrow[r]);
    }
  }
}

// ---------------- host-side orchestration ----------------
extern "C" void kernel_launch(void* const* d_in, const int* in_sizes, int n_in,
                              void* d_out, int out_size, void* d_ws, size_t ws_size,
                              hipStream_t stream)
{
  (void)in_sizes; (void)n_in; (void)out_size; (void)ws_size;
  const float* x    = (const float*)d_in[0];
  const float* fcos = (const float*)d_in[1];
  const float* fsin = (const float*)d_in[2];
  const float* wq   = (const float*)d_in[4];
  const float* wk   = (const float*)d_in[5];
  const float* wv   = (const float*)d_in[6];
  const float* wo   = (const float*)d_in[7];
  // d_in[3] mask (causal, applied analytically), d_in[8..10] caches/start_pos unused.

  char* ws = (char*)d_ws;
  bf16* xb  = (bf16*)ws; ws += (size_t)MTOT * DIM * 2;  // 16 MB
  bf16* wqT = (bf16*)ws; ws += (size_t)DIM * DIM * 2;   //  8 MB
  bf16* wkT = (bf16*)ws; ws += (size_t)KVD * DIM * 2;   //  2 MB
  bf16* wvT = (bf16*)ws; ws += (size_t)KVD * DIM * 2;   //  2 MB
  bf16* woT = (bf16*)ws; ws += (size_t)DIM * DIM * 2;   //  8 MB
  bf16* qb  = (bf16*)ws; ws += (size_t)MTOT * DIM * 2;  // 16 MB
  bf16* kb  = (bf16*)ws; ws += (size_t)MTOT * KVD * 2;  //  4 MB
  bf16* vt  = (bf16*)ws; ws += (size_t)KVD * MTOT * 2;  //  4 MB
  bf16* ob  = (bf16*)ws; ws += (size_t)MTOT * DIM * 2;  // 16 MB  (total 76 MB)

  const int nx = MTOT * DIM;
  cvt_f32_bf16 <<<(nx + 255) / 256, 256, 0, stream>>>(x, xb, nx);
  cvtT_f32_bf16<<<(DIM * DIM + 255) / 256, 256, 0, stream>>>(wq, wqT, DIM, DIM);
  cvtT_f32_bf16<<<(DIM * KVD + 255) / 256, 256, 0, stream>>>(wk, wkT, DIM, KVD);
  cvtT_f32_bf16<<<(DIM * KVD + 255) / 256, 256, 0, stream>>>(wv, wvT, DIM, KVD);
  cvtT_f32_bf16<<<(DIM * DIM + 255) / 256, 256, 0, stream>>>(wo, woT, DIM, DIM);

  // Q/K projections with fused rotary; V projection stored transposed
  gemm_bf16_kernel<<<dim3(MTOT / 32, DIM / 512), 256, 0, stream>>>(
      xb, wqT, qb, DIM, DIM, GM_ROPE, fcos, fsin);
  gemm_bf16_kernel<<<dim3(MTOT / 32, KVD / 512), 256, 0, stream>>>(
      xb, wkT, kb, DIM, KVD, GM_ROPE, fcos, fsin);
  gemm_bf16_kernel<<<dim3(MTOT / 32, KVD / 512), 256, 0, stream>>>(
      xb, wvT, vt, DIM, KVD, GM_VT, nullptr, nullptr);

  // causal flash attention: 2*32*128 = 8192 waves -> 1024 blocks
  attn_fwd_kernel<<<(BATCH * NH * (SEQ / 16)) / 8, 256, 0, stream>>>(qb, kb, vt, ob);

  // output projection -> f32 result
  gemm_bf16_kernel<<<dim3(MTOT / 32, DIM / 512), 256, 0, stream>>>(
      ob, woT, d_out, DIM, DIM, GM_F32, nullptr, nullptr);
}
